// MLP_11871289606695
// MI455X (gfx1250) — matrix operand order repaired and checked by reading, not compiled, not run on hardware
//
#include <hip/hip_runtime.h>
#include <math.h>

// ---------------------------------------------------------------------------
// Types for CDNA5 WMMA (wave32)
// ---------------------------------------------------------------------------
typedef __attribute__((ext_vector_type(16))) __bf16 v16bf;
typedef __attribute__((ext_vector_type(8)))  __bf16 v8bf;
typedef __attribute__((ext_vector_type(4)))  __bf16 v4bf;
typedef __attribute__((ext_vector_type(8)))  float  v8f;
typedef __attribute__((ext_vector_type(4)))  float  v4f;
typedef __attribute__((ext_vector_type(4)))  int    v4i;

// Problem constants (match the reference)
constexpr int NROWS = 65536;
constexpr int DH    = 256;   // D_IN == D_H == 256
constexpr int DOUT  = 128;
constexpr int NEXP  = 16;

// GEMM tiling
constexpr int TM = 128;   // rows per block
constexpr int TN = 128;   // cols per block
constexpr int KC = 32;    // K chunk per WMMA step (bf16 16x16x32)

// ---------------------------------------------------------------------------
// gfx1250 async global->LDS copies (ASYNCcnt path), with safe fallback
// ---------------------------------------------------------------------------
#if defined(__HIP_DEVICE_COMPILE__) && __has_builtin(__builtin_amdgcn_global_load_async_to_lds_b128)
#define ASYNC_LDS 1
#else
#define ASYNC_LDS 0
#endif

__device__ __forceinline__ void wait_async_zero() {
#if ASYNC_LDS
#if __has_builtin(__builtin_amdgcn_s_wait_asynccnt)
  __builtin_amdgcn_s_wait_asynccnt(0);
#else
  asm volatile("s_wait_asynccnt 0x0" ::: "memory");
#endif
#endif
}

#if ASYNC_LDS
typedef __attribute__((address_space(1))) v4i as1_v4i;   // global (prints as __device__)
typedef __attribute__((address_space(3))) v4i as3_v4i;   // LDS    (prints as __shared__)
__device__ __forceinline__ void async_copy_b128(const void* g, void* l) {
  __builtin_amdgcn_global_load_async_to_lds_b128((as1_v4i*)g, (as3_v4i*)l, 0, 0);
}
#endif

// ---------------------------------------------------------------------------
// Weight transpose + f32->bf16 convert:  Wt[mat][n][k] = (bf16) W[mat][k][n]
// ---------------------------------------------------------------------------
__global__ __launch_bounds__(256)
void tconv_kernel(const float* __restrict__ W, __bf16* __restrict__ Wt,
                  int K, int Nn, int total) {
  int idx = blockIdx.x * 256 + threadIdx.x;
  if (idx >= total) return;
  int kn  = K * Nn;
  int mat = idx / kn;
  int rem = idx - mat * kn;
  int k = rem / Nn;
  int n = rem - k * Nn;
  Wt[(size_t)mat * kn + (size_t)n * K + k] = (__bf16)W[idx];
}

// ---------------------------------------------------------------------------
// Zero helpers
// ---------------------------------------------------------------------------
__global__ __launch_bounds__(256)
void zero_f32_kernel(float* __restrict__ p, long long count) {
  long long idx = (long long)blockIdx.x * 256 + threadIdx.x;
  if (idx < count) p[idx] = 0.0f;
}

__global__ __launch_bounds__(256)
void zero_route_kernel(int* __restrict__ cnt, float* __restrict__ partials) {
  if (threadIdx.x < NEXP) cnt[threadIdx.x] = 0;
  partials[threadIdx.x] = 0.0f;
}

// ---------------------------------------------------------------------------
// Gate: logits = h@Wg + bg; full softmax -> per-row std (ddof=1), block-reduced
// deterministically; top-2 -> renormalized pair; routing lists via atomics.
// If yin != nullptr, h is produced on the fly as relu(yin) and stored to hout
// (fuses the former combine pass for MoE layer 2).
// ---------------------------------------------------------------------------
__global__ __launch_bounds__(256)
void gate_kernel(const __bf16* __restrict__ h, const float* __restrict__ yin,
                 __bf16* __restrict__ hout, const float* __restrict__ Wg,
                 const float* __restrict__ bg, int* __restrict__ cnt,
                 int* __restrict__ rowlist, float* __restrict__ wlist,
                 float* __restrict__ partials) {
  __shared__ float red[256];
  const int r = blockIdx.x * 256 + threadIdx.x;

  float lg[NEXP];
#pragma unroll
  for (int e = 0; e < NEXP; ++e) lg[e] = bg[e];

  if (yin) {
    // fused: h = relu(y) -> bf16, and accumulate gate logits from relu(y)
    const float* yr = yin + (size_t)r * DH;
    __bf16* hw = hout + (size_t)r * DH;
    for (int k = 0; k < DH; k += 4) {
      v4f f = *(const v4f*)(yr + k);
      v4bf o;
#pragma unroll
      for (int j = 0; j < 4; ++j) {
        float v = fmaxf(f[j], 0.0f);
        o[j] = (__bf16)v;
        const float* wr = Wg + (k + j) * NEXP;
#pragma unroll
        for (int e = 0; e < NEXP; ++e) lg[e] = fmaf(v, wr[e], lg[e]);
      }
      *(v4bf*)(hw + k) = o;
    }
  } else {
    const __bf16* hr = h + (size_t)r * DH;
    for (int k = 0; k < DH; k += 8) {
      v8bf hv8 = *(const v8bf*)(hr + k);
#pragma unroll
      for (int j = 0; j < 8; ++j) {
        float hv = (float)hv8[j];
        const float* wr = Wg + (k + j) * NEXP;
#pragma unroll
        for (int e = 0; e < NEXP; ++e) lg[e] = fmaf(hv, wr[e], lg[e]);
      }
    }
  }

  // full softmax (for the std statistic)
  float mx = lg[0];
#pragma unroll
  for (int e = 1; e < NEXP; ++e) mx = fmaxf(mx, lg[e]);
  float p[NEXP]; float s = 0.0f;
#pragma unroll
  for (int e = 0; e < NEXP; ++e) { p[e] = __expf(lg[e] - mx); s += p[e]; }
  float inv = 1.0f / s;
  float mean = 0.0f;
#pragma unroll
  for (int e = 0; e < NEXP; ++e) { p[e] *= inv; mean += p[e]; }
  mean *= (1.0f / NEXP);
  float var = 0.0f;
#pragma unroll
  for (int e = 0; e < NEXP; ++e) { float d = p[e] - mean; var = fmaf(d, d, var); }
  float sd = sqrtf(var * (1.0f / (NEXP - 1)));   // unbiased (ddof=1)

  // deterministic block reduction of std
  red[threadIdx.x] = sd;
  __syncthreads();
  for (int off = 128; off > 0; off >>= 1) {
    if (threadIdx.x < off) red[threadIdx.x] += red[threadIdx.x + off];
    __syncthreads();
  }
  if (threadIdx.x == 0) partials[blockIdx.x] = red[0];

  // top-2 (stable: first index wins ties, like lax.top_k)
  float v0 = -1e30f; int e0 = 0;
#pragma unroll
  for (int e = 0; e < NEXP; ++e) if (lg[e] > v0) { v0 = lg[e]; e0 = e; }
  float v1 = -1e30f; int e1 = 0;
#pragma unroll
  for (int e = 0; e < NEXP; ++e) if (e != e0 && lg[e] > v1) { v1 = lg[e]; e1 = e; }
  float s0 = 1.0f / (1.0f + __expf(v1 - v0));
  float s1 = 1.0f - s0;

  int p0 = atomicAdd(&cnt[e0], 1);
  rowlist[e0 * NROWS + p0] = r; wlist[e0 * NROWS + p0] = s0;
  int p1 = atomicAdd(&cnt[e1], 1);
  rowlist[e1 * NROWS + p1] = r; wlist[e1 * NROWS + p1] = s1;
}

// ---------------------------------------------------------------------------
// WMMA GEMM, 128x128 tile, K=256 in chunks of 32, bf16 A/B, f32 accum.
// Double-buffered LDS tiles; async global->LDS (ASYNCcnt) when available;
// all 8 B fragments register-blocked so the 8 WMMAs issue back-to-back.
//  MODE 0: A = x (f32, sequential rows)    -> C = relu(A@W0+b0) -> bf16 hbuf
//  MODE 1: A = h (bf16, gathered rows)     -> w*(A@We+be) atomically into f32 y
//  MODE 2: A = relu(y) (f32, sequential)   -> C = A@Wf+bf -> f32 out
// ---------------------------------------------------------------------------
template <int MODE>
__global__ __launch_bounds__(256)
void gemm_kernel(const float* __restrict__ Af32, const __bf16* __restrict__ Abf,
                 const __bf16* __restrict__ Bt, const float* __restrict__ bias,
                 const int* __restrict__ cnt, const int* __restrict__ rowlist,
                 const float* __restrict__ wlist, float* __restrict__ Yatomic,
                 __bf16* __restrict__ Hout, float* __restrict__ Fout,
                 int outStride) {
  __shared__ __bf16 As[2][TM][KC];
  __shared__ __bf16 Bs[2][TN][KC];
  __shared__ int    rows_s[TM];
  __shared__ float  w_s[TM];

  const int tid  = threadIdx.x;
  const int wv   = tid >> 5;
  const int lane = tid & 31;
  const int n0   = blockIdx.z * TN;
  const int t0   = blockIdx.x * TM;

  int ecnt = 0;
  const __bf16* Bte = Bt;
  const float*  be  = bias;
  if constexpr (MODE == 1) {
    const int e = blockIdx.y;
    ecnt = cnt[e];                       // scalar -> uniform early exit
    if (t0 >= ecnt) return;
    Bte = Bt + (size_t)e * DH * DH;
    be  = bias + e * DH;
    if (tid < TM) {
      int idx = t0 + tid;
      int ci  = idx < ecnt ? idx : (ecnt - 1);
      rows_s[tid] = rowlist[e * NROWS + ci];
      w_s[tid]    = (idx < ecnt) ? wlist[e * NROWS + ci] : 0.0f;
    }
    __syncthreads();                     // rows_s visible to all loaders
  }

  const int rl  = tid >> 1;          // 0..127: tile row (A) / tile col (B)
  const int seg = (tid & 1) * 16;    // 16 bf16 elements per thread per tile

  const float*  af32row = nullptr;
  const __bf16* abfrow  = nullptr;
  if constexpr (MODE == 1) abfrow = Abf + (size_t)rows_s[rl] * DH;
  else                     af32row = Af32 + (size_t)(t0 + rl) * DH;
  const __bf16* btrow = Bte + (size_t)(n0 + rl) * DH;

  // ---- tile movers -------------------------------------------------------
  auto issueB = [&](int kc, int buf) {
    const __bf16* s = btrow + kc + seg;
#if ASYNC_LDS
    async_copy_b128(s,     &Bs[buf][rl][seg]);
    async_copy_b128(s + 8, &Bs[buf][rl][seg + 8]);
#else
    *(v8bf*)&Bs[buf][rl][seg]     = *(const v8bf*)s;
    *(v8bf*)&Bs[buf][rl][seg + 8] = *(const v8bf*)(s + 8);
#endif
  };
  auto issueA1 = [&](int kc, int buf) {       // MODE 1: bf16 gathered rows
    const __bf16* s = abfrow + kc + seg;
#if ASYNC_LDS
    async_copy_b128(s,     &As[buf][rl][seg]);
    async_copy_b128(s + 8, &As[buf][rl][seg + 8]);
#else
    *(v8bf*)&As[buf][rl][seg]     = *(const v8bf*)s;
    *(v8bf*)&As[buf][rl][seg + 8] = *(const v8bf*)(s + 8);
#endif
  };
  v4f st[4];                                  // f32 staging (modes 0/2)
  auto loadA32 = [&](int kc) {
#pragma unroll
    for (int j = 0; j < 4; ++j) st[j] = *(const v4f*)(af32row + kc + seg + j * 4);
  };
  auto storeA32 = [&](int buf) {
    v8bf lo, hi;
#pragma unroll
    for (int j = 0; j < 4; ++j) {
#pragma unroll
      for (int i = 0; i < 4; ++i) {
        float v = st[j][i];
        if constexpr (MODE == 2) v = fmaxf(v, 0.0f);   // relu(y) on the fly
        if (j < 2) lo[j * 4 + i] = (__bf16)v;
        else       hi[(j - 2) * 4 + i] = (__bf16)v;
      }
    }
    *(v8bf*)&As[buf][rl][seg]     = lo;
    *(v8bf*)&As[buf][rl][seg + 8] = hi;
  };

  // ---- accumulators ------------------------------------------------------
  v8f acc[8];
#pragma unroll
  for (int t = 0; t < 8; ++t)
#pragma unroll
    for (int j = 0; j < 8; ++j) acc[t][j] = 0.0f;

  auto compute = [&](int buf) {
    union F { v16bf v; v8bf h[2]; } afr, bfr[8];
    const int mrow = wv * 16 + (lane & 15);
    const int b1   = (lane & 16) ? 8 : 0;     // A K-group select
    const int k0   = (lane & 16) ? 16 : 0;    // unused; B reads use b1 too
    afr.h[0] = *(const v8bf*)&As[buf][mrow][b1];
    afr.h[1] = *(const v8bf*)&As[buf][mrow][16 + b1];
#pragma unroll
    for (int nt = 0; nt < 8; ++nt) {          // register-block all B frags
      const int nrow = nt * 16 + (lane & 15);
      bfr[nt].h[0] = *(const v8bf*)&Bs[buf][nrow][b1];
      bfr[nt].h[1] = *(const v8bf*)&Bs[buf][nrow][16 + b1];
    }
#pragma unroll
    for (int nt = 0; nt < 8; ++nt)
      acc[nt] = __builtin_amdgcn_wmma_f32_16x16x32_bf16(
          false, afr.v, false, bfr[nt].v, (short)0, acc[nt], false, false);
  };

  // ---- software-pipelined main loop --------------------------------------
  issueB(0, 0);
  if constexpr (MODE == 1) issueA1(0, 0);
  else { loadA32(0); storeA32(0); }

  for (int kc = 0; kc < DH; kc += KC) {
    const int cur = (kc >> 5) & 1;
    wait_async_zero();
    __syncthreads();                         // buffer `cur` ready
    const bool more = (kc + KC) < DH;
    if (more) {
      issueB(kc + KC, cur ^ 1);
      if constexpr (MODE == 1) issueA1(kc + KC, cur ^ 1);
      else loadA32(kc + KC);                 // global f32 in flight over WMMA
    }
    compute(cur);
    if (more) {
      if constexpr (MODE != 1) storeA32(cur ^ 1);
    }
  }

  // ---- epilogue ----------------------------------------------------------
  const int nlane = lane & 15;
  const int mhi   = (lane & 16) ? 8 : 0;
#pragma unroll
  for (int nt = 0; nt < 8; ++nt) {
    const int n  = n0 + nt * 16 + nlane;
    const float bv = be[n];
#pragma unroll
    for (int v = 0; v < 8; ++v) {
      const int ml = wv * 16 + v + mhi;
      float val = acc[nt][v] + bv;
      if constexpr (MODE == 0) {
        val = fmaxf(val, 0.0f);
        Hout[(size_t)(t0 + ml) * outStride + n] = (__bf16)val;
      } else if constexpr (MODE == 1) {
        if (t0 + ml < ecnt) {
          const int row = rows_s[ml];
          atomicAdd(&Yatomic[(size_t)row * outStride + n], val * w_s[ml]);
        }
      } else {
        Fout[(size_t)(t0 + ml) * outStride + n] = val;
      }
    }
  }
}

// ---------------------------------------------------------------------------
// scalar output: (mean(std1) + mean(std2)) * 0.5
// ---------------------------------------------------------------------------
__global__ __launch_bounds__(256)
void finalize_kernel(const float* __restrict__ p1, const float* __restrict__ p2,
                     float* __restrict__ out) {
  __shared__ float red[256];
  red[threadIdx.x] = p1[threadIdx.x] + p2[threadIdx.x];
  __syncthreads();
  for (int off = 128; off > 0; off >>= 1) {
    if (threadIdx.x < off) red[threadIdx.x] += red[threadIdx.x + off];
    __syncthreads();
  }
  if (threadIdx.x == 0) out[0] = red[0] * (0.5f / (float)NROWS);
}

// ---------------------------------------------------------------------------
extern "C" void kernel_launch(void* const* d_in, const int* in_sizes, int n_in,
                              void* d_out, int out_size, void* d_ws, size_t ws_size,
                              hipStream_t stream) {
  const float* x   = (const float*)d_in[0];
  const float* W0  = (const float*)d_in[1];
  const float* b0  = (const float*)d_in[2];
  const float* Wg1 = (const float*)d_in[3];
  const float* bg1 = (const float*)d_in[4];
  const float* We1 = (const float*)d_in[5];
  const float* be1 = (const float*)d_in[6];
  const float* Wg2 = (const float*)d_in[7];
  const float* bg2 = (const float*)d_in[8];
  const float* We2 = (const float*)d_in[9];
  const float* be2 = (const float*)d_in[10];
  const float* Wf  = (const float*)d_in[11];
  const float* bfv = (const float*)d_in[12];
  float* out = (float*)d_out;

  char* ws = (char*)d_ws;
  size_t off = 0;
  auto alloc = [&](size_t bytes) -> void* {
    void* p = ws + off;
    off = (off + bytes + 255) & ~(size_t)255;
    return p;
  };
  __bf16* W0t  = (__bf16*)alloc((size_t)DH * DH * 2);
  __bf16* We1t = (__bf16*)alloc((size_t)NEXP * DH * DH * 2);
  __bf16* We2t = (__bf16*)alloc((size_t)NEXP * DH * DH * 2);
  __bf16* Wft  = (__bf16*)alloc((size_t)DOUT * DH * 2);
  __bf16* hbuf = (__bf16*)alloc((size_t)NROWS * DH * 2);
  float*  y    = (float*)alloc((size_t)NROWS * DH * 4);
  int*    cnt  = (int*)alloc(NEXP * 4);
  int*    rowl = (int*)alloc((size_t)NEXP * NROWS * 4);
  float*  wl   = (float*)alloc((size_t)NEXP * NROWS * 4);
  float*  par1 = (float*)alloc(256 * 4);
  float*  par2 = (float*)alloc(256 * 4);
  (void)ws_size; (void)in_sizes; (void)n_in; (void)out_size;

  const dim3 blk(256);
  const long long ycount = (long long)NROWS * DH;

  // 0) weights -> transposed bf16
  tconv_kernel<<<(DH * DH + 255) / 256, blk, 0, stream>>>(W0, W0t, DH, DH, DH * DH);
  tconv_kernel<<<(NEXP * DH * DH + 255) / 256, blk, 0, stream>>>(We1, We1t, DH, DH, NEXP * DH * DH);
  tconv_kernel<<<(NEXP * DH * DH + 255) / 256, blk, 0, stream>>>(We2, We2t, DH, DH, NEXP * DH * DH);
  tconv_kernel<<<(DH * DOUT + 255) / 256, blk, 0, stream>>>(Wf, Wft, DH, DOUT, DH * DOUT);

  // 1) h1 = relu(x @ W0 + b0) -> hbuf (bf16)
  gemm_kernel<0><<<dim3(NROWS / TM, 1, DH / TN), blk, 0, stream>>>(
      x, nullptr, W0t, b0, nullptr, nullptr, nullptr, nullptr, hbuf, nullptr, DH);

  // 2) MoE layer 1
  zero_f32_kernel<<<(unsigned)(ycount / 256), blk, 0, stream>>>(y, ycount);
  zero_route_kernel<<<1, blk, 0, stream>>>(cnt, par1);
  gate_kernel<<<NROWS / 256, blk, 0, stream>>>(hbuf, nullptr, nullptr, Wg1, bg1,
                                               cnt, rowl, wl, par1);
  gemm_kernel<1><<<dim3(NROWS / TM, NEXP, DH / TN), blk, 0, stream>>>(
      nullptr, hbuf, We1t, be1, cnt, rowl, wl, y, nullptr, nullptr, DH);

  // 3) MoE layer 2 (gate fused with relu(y1) -> h2)
  zero_route_kernel<<<1, blk, 0, stream>>>(cnt, par2);
  gate_kernel<<<NROWS / 256, blk, 0, stream>>>(nullptr, y, hbuf, Wg2, bg2,
                                               cnt, rowl, wl, par2);
  zero_f32_kernel<<<(unsigned)(ycount / 256), blk, 0, stream>>>(y, ycount);
  gemm_kernel<1><<<dim3(NROWS / TM, NEXP, DH / TN), blk, 0, stream>>>(
      nullptr, hbuf, We2t, be2, cnt, rowl, wl, y, nullptr, nullptr, DH);

  // 4) out = relu(y2) @ Wf + bf (relu fused into the A-tile stage)
  gemm_kernel<2><<<dim3(NROWS / TM, 1, 1), blk, 0, stream>>>(
      y, nullptr, Wft, bfv, nullptr, nullptr, nullptr, nullptr, nullptr, out, DOUT);

  // 5) scalar gate statistic
  finalize_kernel<<<1, blk, 0, stream>>>(par1, par2, out + (size_t)NROWS * DOUT);
}
